// Encoder_22402549416572
// MI455X (gfx1250) — matrix operand order repaired and checked by reading, not compiled, not run on hardware
//
#include <hip/hip_runtime.h>
#include <hip/hip_bf16.h>
#include <math.h>

// ---------------------------------------------------------------------------
// GAT encoder for MI455X (gfx1250, wave32).
// GEMMs: A pre-converted to f16 (K padded to 32), W pre-swizzled to WMMA
// fragment order in global memory. Inner loop per wave per K-tile:
//   1x global_load_async_to_lds_b128 (A stage, ASYNCcnt path; VGPR-staged
//   fallback if builtin unavailable) + 2x global b128 (B prefetch)
//   + 8x ds_load_b128 (A frags) + 4x v_wmma_f32_16x16x32_f16.
// Edge softmax/aggregation via float atomics (segment max/sum).
// ---------------------------------------------------------------------------

typedef __attribute__((ext_vector_type(16))) _Float16 v16h;
typedef __attribute__((ext_vector_type(8)))  _Float16 v8h;
typedef __attribute__((ext_vector_type(2)))  _Float16 h2;
typedef __attribute__((ext_vector_type(8)))  float    v8f;
typedef __attribute__((ext_vector_type(4)))  int      v4i;

#if defined(__has_builtin)
#if __has_builtin(__builtin_amdgcn_global_load_async_to_lds_b128) && \
    __has_builtin(__builtin_amdgcn_s_wait_asynccnt)
#define HAVE_ASYNC_LDS 1
#endif
#endif

typedef __attribute__((address_space(3))) v4i* lds_v4i_t;

#define NNODES 50000
#define NEDGES 800000

#define AS 40   // A tile row stride in halves (80B rows: 16B-aligned b128 ops)

// Monotonic float<->uint encoding so atomicMax(uint) == float max.
__device__ __forceinline__ unsigned enc_f32(float v) {
  unsigned b = __float_as_uint(v);
  return (b & 0x80000000u) ? ~b : (b | 0x80000000u);
}
__device__ __forceinline__ float dec_f32(unsigned e) {
  return (e & 0x80000000u) ? __uint_as_float(e & 0x7FFFFFFFu)
                           : __uint_as_float(~e);
}
__device__ __forceinline__ float lrelu(float v) { return v > 0.f ? v : 0.2f * v; }

// ---------------------------------------------------------------------------
// f32 [n,K] -> f16 [n,Kpad], zero-padded K..Kpad. K even, Kpad mult of 32.
// ---------------------------------------------------------------------------
__global__ void cvt_pad_f16_kernel(const float* __restrict__ in,
                                   _Float16* __restrict__ out,
                                   int n, int K, int Kpad) {
  int ppr = Kpad >> 1;
  int i = blockIdx.x * blockDim.x + threadIdx.x;
  if (i >= n * ppr) return;
  int row = i / ppr, c = (i % ppr) << 1;
  h2 hv; hv[0] = (_Float16)0.f; hv[1] = (_Float16)0.f;
  if (c < K) {
    float2 v = *(const float2*)(in + (size_t)row * K + c);
    hv[0] = (_Float16)v.x; hv[1] = (_Float16)v.y;
  }
  *(h2*)(out + (size_t)row * Kpad + c) = hv;
}

// ---------------------------------------------------------------------------
// W [K,Ncols] f32 -> Wf fragment-order f16: [Kpad/32][Ncols/16][32 lanes][16].
// Lane l of col-tile ct holds col ct*16+(l&15), k = kt*32 + (l>>4)*16 + idx.
// ---------------------------------------------------------------------------
__global__ void wfrag_kernel(const float* __restrict__ W,
                             _Float16* __restrict__ Wf,
                             int K, int Kpad, int Ncols) {
  int total = (Kpad * Ncols) >> 1;
  int i = blockIdx.x * blockDim.x + threadIdx.x;
  if (i >= total) return;
  int lin = i << 1;                    // half offset in Wf
  int idx  = lin & 15;
  int lane = (lin >> 4) & 31;
  int ctkt = lin >> 9;
  int NT = Ncols >> 4;
  int ct = ctkt % NT, kt = ctkt / NT;
  int k   = (kt << 5) + ((lane >> 4) << 4) + idx;   // even
  int col = (ct << 4) + (lane & 15);
  h2 hv; hv[0] = (_Float16)0.f; hv[1] = (_Float16)0.f;
  if (k < K) {                                      // K even: k+1 < K too
    hv[0] = (_Float16)W[(size_t)k * Ncols + col];
    hv[1] = (_Float16)W[(size_t)(k + 1) * Ncols + col];
  }
  *(h2*)(Wf + lin) = hv;
}

// ---------------------------------------------------------------------------
// C[M,Ncols](f32) = A[M,K](f16) @ Wf(frag-order f16).  K mult of 32.
// Block: 128 thr = 4 waves; tile 64 rows x 64 cols; wave w: cols 16w..16w+15,
// 4 row-block accumulators. Double-buffered LDS A tile (async copy when
// available), B direct from global prefetched one tile ahead.
// ---------------------------------------------------------------------------
__global__ __launch_bounds__(128) void gemm_wmma_f16(
    const _Float16* __restrict__ A, const _Float16* __restrict__ Wf,
    float* __restrict__ C, int M, int K, int Ncols) {
  __shared__ __align__(16) _Float16 As[2][64][AS];

  const int tid  = threadIdx.x;
  const int wave = tid >> 5;
  const int lane = tid & 31;
  const int row0 = blockIdx.x * 64;
  const int colB = blockIdx.y * 64;
  const int hsel = lane >> 4;
  const int mrow = lane & 15;
  const int kbA  = hsel * 16;

  const int NT = Ncols >> 4;
  int ct = (colB >> 4) + wave; ct = ct < NT ? ct : NT - 1;    // clamp

  // A staging: thread -> row rA = tid>>1, half-col cA = (tid&1)*16 (16B each)
  const int rA = tid >> 1, cA = (tid & 1) << 4;
  int rowA = row0 + rA; rowA = rowA < M ? rowA : M - 1;       // clamp
  const _Float16* Aptr = A + (size_t)rowA * K + cA;
  _Float16* const ldsA = &As[0][0][0] + (rA * AS + cA);
  const int bufStride = 64 * AS;                              // halves

  const int ktiles = K >> 5;
  v8f acc[4] = {};

#if defined(HAVE_ASYNC_LDS)
  // ---- async global->LDS staging (no VGPR round trip, ASYNCcnt) ----
  __builtin_amdgcn_global_load_async_to_lds_b128(
      (v4i*)Aptr, (lds_v4i_t)ldsA, 0, 0);
  v16h bv = *(const v16h*)(Wf + (size_t)((ct << 5) + lane) * 16);

  for (int kt = 0; kt < ktiles; ++kt) {
    const int buf = kt & 1;
    __builtin_amdgcn_s_wait_asynccnt(0);    // own tile landed in LDS
    __syncthreads();                        // everyone's tile landed
    v16h bvn = bv;
    if (kt + 1 < ktiles) {                  // prefetch next tile
      __builtin_amdgcn_global_load_async_to_lds_b128(
          (v4i*)(Aptr + ((kt + 1) << 5)),
          (lds_v4i_t)(ldsA + (buf ^ 1) * bufStride), 0, 0);
      bvn = *(const v16h*)(Wf + (size_t)((((kt + 1) * NT + ct) << 5) + lane) * 16);
    }
    #pragma unroll
    for (int rb = 0; rb < 4; ++rb) {
      const _Float16* ap = &As[buf][rb * 16 + mrow][0];
      v8h alo = *(const v8h*)(ap + kbA);
      v8h ahi = *(const v8h*)(ap + kbA + 8);
      v16h a = __builtin_shufflevector(alo, ahi,
                 0,1,2,3,4,5,6,7,8,9,10,11,12,13,14,15);
      acc[rb] = __builtin_amdgcn_wmma_f32_16x16x32_f16(
          false, a, false, bv, (short)0, acc[rb], false, false);
    }
    bv = bvn;
    // one barrier/tile is safe: reads(kt-1) precede barrier(kt) precede
    // async-write(kt) to the opposite buffer.
  }
#else
  // ---- fallback: VGPR-staged double-buffered copy ----
  v8h  av = *(const v8h*)(Aptr);
  v16h bv = *(const v16h*)(Wf + (size_t)((ct << 5) + lane) * 16);

  for (int kt = 0; kt < ktiles; ++kt) {
    const int buf = kt & 1;
    *(v8h*)(ldsA + buf * bufStride) = av;
    __syncthreads();
    v8h avn = av; v16h bvn = bv;
    if (kt + 1 < ktiles) {                    // prefetch next tile
      avn = *(const v8h*)(Aptr + ((kt + 1) << 5));
      bvn = *(const v16h*)(Wf + (size_t)((((kt + 1) * NT + ct) << 5) + lane) * 16);
    }
    #pragma unroll
    for (int rb = 0; rb < 4; ++rb) {
      const _Float16* ap = &As[buf][rb * 16 + mrow][0];
      v8h alo = *(const v8h*)(ap + kbA);
      v8h ahi = *(const v8h*)(ap + kbA + 8);
      v16h a = __builtin_shufflevector(alo, ahi,
                 0,1,2,3,4,5,6,7,8,9,10,11,12,13,14,15);
      acc[rb] = __builtin_amdgcn_wmma_f32_16x16x32_f16(
          false, a, false, bv, (short)0, acc[rb], false, false);
    }
    av = avn; bv = bvn;
  }
#endif

  const int col = colB + (wave << 4) + mrow;
  if (col < Ncols) {
    if (row0 + 64 <= M) {
      #pragma unroll
      for (int rb = 0; rb < 4; ++rb)
        #pragma unroll
        for (int r = 0; r < 8; ++r) {
          int row = row0 + rb * 16 + hsel * 8 + r;
          C[(size_t)row * Ncols + col] = acc[rb][r];
        }
    } else {
      #pragma unroll
      for (int rb = 0; rb < 4; ++rb)
        #pragma unroll
        for (int r = 0; r < 8; ++r) {
          int row = row0 + rb * 16 + hsel * 8 + r;
          if (row < M) C[(size_t)row * Ncols + col] = acc[rb][r];
        }
    }
  }
}

// el[n,h] = sum_f feat[n,h,f]*al[h,f]; er likewise. F multiple of 4.
__global__ void scores_kernel(const float* __restrict__ feat,
                              const float* __restrict__ al,
                              const float* __restrict__ ar,
                              float* __restrict__ el, float* __restrict__ er,
                              int n, int H, int F) {
  int i = blockIdx.x * blockDim.x + threadIdx.x;
  if (i >= n * H) return;
  int node = i / H, h = i % H;
  const float4* fp = (const float4*)(feat + (size_t)node * H * F + (size_t)h * F);
  const float4* alp = (const float4*)(al + h * F);
  const float4* arp = (const float4*)(ar + h * F);
  float sl = 0.f, sr = 0.f;
  for (int f = 0; f < (F >> 2); ++f) {
    float4 v = fp[f], l4 = alp[f], r4 = arp[f];
    sl += v.x * l4.x + v.y * l4.y + v.z * l4.z + v.w * l4.w;
    sr += v.x * r4.x + v.y * r4.y + v.z * r4.z + v.w * r4.w;
  }
  el[i] = sl;
  er[i] = sr;
}

__global__ void edge_max_kernel(const int* __restrict__ src, const int* __restrict__ dst,
                                const float* __restrict__ el, const float* __restrict__ er,
                                unsigned* __restrict__ emax, int E, int H) {
  int i = blockIdx.x * blockDim.x + threadIdx.x;
  if (i >= E * H) return;
  int e = i / H, h = i % H;
  int s = src[e], d = dst[e];
  float v = lrelu(el[s * H + h] + er[d * H + h]);
  atomicMax(&emax[d * H + h], enc_f32(v));
}

__global__ void edge_denom_kernel(const int* __restrict__ src, const int* __restrict__ dst,
                                  const float* __restrict__ el, const float* __restrict__ er,
                                  const unsigned* __restrict__ emax,
                                  float* __restrict__ denom, int E, int H) {
  int i = blockIdx.x * blockDim.x + threadIdx.x;
  if (i >= E * H) return;
  int e = i / H, h = i % H;
  int s = src[e], d = dst[e];
  float v = lrelu(el[s * H + h] + er[d * H + h]);
  atomicAdd(&denom[d * H + h], expf(v - dec_f32(emax[d * H + h])));
}

// One wave per edge: agg[dst,:] += alpha[h] * feat[src,:]
__global__ __launch_bounds__(128) void edge_agg_kernel(
    const int* __restrict__ src, const int* __restrict__ dst,
    const float* __restrict__ el, const float* __restrict__ er,
    const unsigned* __restrict__ emax, const float* __restrict__ denom,
    const float* __restrict__ feat, float* __restrict__ agg,
    int E, int H, int F) {
  int e = blockIdx.x * 4 + (threadIdx.x >> 5);
  if (e >= E) return;
  int lane = threadIdx.x & 31;
  int s = src[e], d = dst[e];
  float alpha[2];
  for (int h = 0; h < H; ++h) {
    float v = lrelu(el[s * H + h] + er[d * H + h]);
    alpha[h] = expf(v - dec_f32(emax[d * H + h])) / denom[d * H + h];
  }
  int HF = H * F;
  for (int c = lane; c < HF; c += 32) {
    int h = c / F;
    atomicAdd(&agg[(size_t)d * HF + c], alpha[h] * feat[(size_t)s * HF + c]);
  }
}

// out[n,f] = mean_h(agg[n,h,f] + bias[h,f]) (+res) (exp if doExp)
// Optional f32 and f16 outputs.
__global__ void finalize_kernel(const float* __restrict__ agg,
                                const float* __restrict__ bias,
                                const float* __restrict__ res,
                                float* __restrict__ out,
                                _Float16* __restrict__ outh,
                                int n, int H, int F, int doExp) {
  int i = blockIdx.x * blockDim.x + threadIdx.x;
  if (i >= n * F) return;
  int node = i / F, f = i % F;
  float acc = 0.f;
  for (int h = 0; h < H; ++h)
    acc += agg[(size_t)node * H * F + (size_t)h * F + f] + bias[h * F + f];
  acc /= (float)H;
  if (res) acc += res[i];
  if (doExp) acc = expf(acc);
  if (out)  out[i]  = acc;
  if (outh) outh[i] = (_Float16)acc;
}

__global__ void repar_kernel(const float* __restrict__ mean, const float* __restrict__ var,
                             const float* __restrict__ eps, float* __restrict__ z, int n) {
  int i = blockIdx.x * blockDim.x + threadIdx.x;
  if (i >= n) return;
  z[i] = mean[i] + sqrtf(var[i]) * eps[i];
}

extern "C" void kernel_launch(void* const* d_in, const int* in_sizes, int n_in,
                              void* d_out, int out_size, void* d_ws, size_t ws_size,
                              hipStream_t stream) {
  const float* x   = (const float*)d_in[0];
  const int*   src = (const int*)d_in[1];
  const int*   dst = (const int*)d_in[2];
  const float* W1  = (const float*)d_in[3];
  const float* al1 = (const float*)d_in[4];
  const float* ar1 = (const float*)d_in[5];
  const float* b1  = (const float*)d_in[6];
  const float* W2  = (const float*)d_in[7];
  const float* al2 = (const float*)d_in[8];
  const float* ar2 = (const float*)d_in[9];
  const float* b2  = (const float*)d_in[10];
  const float* Wm  = (const float*)d_in[11];
  const float* alm = (const float*)d_in[12];
  const float* arm = (const float*)d_in[13];
  const float* bm  = (const float*)d_in[14];
  const float* Wv  = (const float*)d_in[15];
  const float* alv = (const float*)d_in[16];
  const float* arv = (const float*)d_in[17];
  const float* bv  = (const float*)d_in[18];
  const float* eps = (const float*)d_in[19];

  const int n = NNODES, e = NEDGES;

  // ---- workspace layout (~284 MB) ----
  float* ws    = (float*)d_ws;
  float* feat1 = ws;                          // n*128 f32 (reused featm/featv)
  float* agg   = feat1 + (size_t)n * 128;     // n*256
  float* o1    = agg   + (size_t)n * 256;     // n*128
  float* feat2 = o1    + (size_t)n * 128;     // n*256
  float* el    = feat2 + (size_t)n * 256;     // n*2
  float* er    = el    + (size_t)n * 2;       // n*2
  unsigned* emax = (unsigned*)(er + (size_t)n * 2);  // n*2
  float* denom = (float*)emax + (size_t)n * 2;       // n*2
  _Float16* xh  = (_Float16*)(denom + (size_t)n * 2); // n*1024
  _Float16* o1h = xh  + (size_t)n * 1024;     // n*128
  _Float16* o2h = o1h + (size_t)n * 128;      // n*128
  _Float16* wf1 = o2h + (size_t)n * 128;      // 1024*128
  _Float16* wf2 = wf1 + (size_t)1024 * 128;   // 128*256
  _Float16* wfm = wf2 + (size_t)128 * 256;    // 128*32
  _Float16* wfv = wfm + (size_t)128 * 32;     // 128*32
  float* featm = feat1;                       // n*32 (feat1 dead by then)
  float* featv = feat1 + (size_t)n * 32;      // n*32

  float* out      = (float*)d_out;            // [z | mean | var], each n*16
  float* z_out    = out;
  float* mean_out = out + (size_t)n * 16;
  float* var_out  = out + (size_t)2 * n * 16;

  auto run_edges = [&](const float* feat, const float* al, const float* ar,
                       float* aggp, int H, int F) {
    int NH = n * H, EH = e * H;
    scores_kernel<<<(NH + 255) / 256, 256, 0, stream>>>(feat, al, ar, el, er, n, H, F);
    (void)hipMemsetAsync(emax, 0, sizeof(unsigned) * (size_t)NH, stream);
    (void)hipMemsetAsync(denom, 0, sizeof(float) * (size_t)NH, stream);
    edge_max_kernel<<<(EH + 255) / 256, 256, 0, stream>>>(src, dst, el, er, emax, e, H);
    edge_denom_kernel<<<(EH + 255) / 256, 256, 0, stream>>>(src, dst, el, er, emax, denom, e, H);
    (void)hipMemsetAsync(aggp, 0, sizeof(float) * (size_t)n * H * F, stream);
    edge_agg_kernel<<<(e + 3) / 4, 128, 0, stream>>>(src, dst, el, er, emax, denom,
                                                     feat, aggp, e, H, F);
  };

  const int mblk = (n + 63) / 64;

  // ---- precompute: f16 inputs + fragment-order weights ----
  cvt_pad_f16_kernel<<<((n * 512) + 255) / 256, 256, 0, stream>>>(x, xh, n, 1000, 1024);
  wfrag_kernel<<<((1024 * 128 / 2) + 255) / 256, 256, 0, stream>>>(W1, wf1, 1000, 1024, 128);
  wfrag_kernel<<<((128 * 256 / 2) + 255) / 256, 256, 0, stream>>>(W2, wf2, 128, 128, 256);
  wfrag_kernel<<<((128 * 32 / 2) + 255) / 256, 256, 0, stream>>>(Wm, wfm, 128, 128, 32);
  wfrag_kernel<<<((128 * 32 / 2) + 255) / 256, 256, 0, stream>>>(Wv, wfv, 128, 128, 32);

  // Layer 1: H=1, F=128, K=1024 (padded)
  gemm_wmma_f16<<<dim3(mblk, 2), 128, 0, stream>>>(xh, wf1, feat1, n, 1024, 128);
  run_edges(feat1, al1, ar1, agg, 1, 128);
  finalize_kernel<<<(n * 128 + 255) / 256, 256, 0, stream>>>(
      agg, b1, nullptr, o1, o1h, n, 1, 128, 0);

  // Layer 2: H=2, F=128, K=128, residual; only f16 output needed downstream
  gemm_wmma_f16<<<dim3(mblk, 4), 128, 0, stream>>>(o1h, wf2, feat2, n, 128, 256);
  run_edges(feat2, al2, ar2, agg, 2, 128);
  finalize_kernel<<<(n * 128 + 255) / 256, 256, 0, stream>>>(
      agg, b2, o1, nullptr, o2h, n, 2, 128, 0);

  // Mean head: H=2, F=16
  gemm_wmma_f16<<<dim3(mblk, 1), 128, 0, stream>>>(o2h, wfm, featm, n, 128, 32);
  run_edges(featm, alm, arm, agg, 2, 16);
  finalize_kernel<<<(n * 16 + 255) / 256, 256, 0, stream>>>(
      agg, bm, nullptr, mean_out, nullptr, n, 2, 16, 0);

  // Var head: H=2, F=16, exp at the end
  gemm_wmma_f16<<<dim3(mblk, 1), 128, 0, stream>>>(o2h, wfv, featv, n, 128, 32);
  run_edges(featv, alv, arv, agg, 2, 16);
  finalize_kernel<<<(n * 16 + 255) / 256, 256, 0, stream>>>(
      agg, bv, nullptr, var_out, nullptr, n, 2, 16, 1);

  // z = mean + sqrt(var) * eps
  repar_kernel<<<(n * 16 + 255) / 256, 256, 0, stream>>>(mean_out, var_out, eps, z_out, n * 16);
}
